// RNNTower_77189152243855
// MI455X (gfx1250) — matrix operand order repaired and checked by reading, not compiled, not run on hardware
//
#include <hip/hip_runtime.h>
#include <math.h>

// ---------------- problem dims ----------------
#define B_   256
#define T_   512
#define E_   128
#define H_   256
#define H3   768   // 3*H

typedef __attribute__((ext_vector_type(16))) __bf16 v16bf;
typedef __attribute__((ext_vector_type(8)))  float  v8f;

union V16 {
    v16bf v;
    __bf16 h[16];
    unsigned short us[16];
    unsigned int   u32[8];
};
union U4 {
    uint4 q;
    unsigned u[4];
    unsigned short s[8];
};

__device__ __forceinline__ unsigned short f2bf(float f) {
    return __builtin_bit_cast(unsigned short, (__bf16)f);   // RNE, hw cvt path
}
__device__ __forceinline__ float bf2f(unsigned short u) {
    return (float)__builtin_bit_cast(__bf16, u);
}
__device__ __forceinline__ unsigned pk_bf16(float lo, float hi) {
    return ((unsigned)f2bf(hi) << 16) | (unsigned)f2bf(lo); // -> v_cvt_pk_bf16_f32
}
__device__ __forceinline__ float sigmoidf_(float x) {
    return 1.0f / (1.0f + __expf(-x));
}
__device__ __forceinline__ float tanhf_(float x) {
    float e = __expf(-2.0f * x);
    return (1.0f - e) / (1.0f + e);
}

// load a 32x16 bf16 B-tile (contiguous 16 bf16 per lane) into a V16
__device__ __forceinline__ void load_btile(V16& bm, const unsigned short* p) {
    const uint4* q = (const uint4*)p;
    uint4 w0 = q[0];
    uint4 w1 = q[1];
    bm.u32[0] = w0.x; bm.u32[1] = w0.y; bm.u32[2] = w0.z; bm.u32[3] = w0.w;
    bm.u32[4] = w1.x; bm.u32[5] = w1.y; bm.u32[6] = w1.z; bm.u32[7] = w1.w;
}

// ---------------- kernel 1: f32 -> bf16 weight conversion ----------------
__global__ __launch_bounds__(512)
void prep_kernel(const float* __restrict__ wihf, const float* __restrict__ whhf,
                 const float* __restrict__ wihb, const float* __restrict__ whhb,
                 unsigned short* __restrict__ wihf_o, unsigned short* __restrict__ whhf_o,
                 unsigned short* __restrict__ wihb_o, unsigned short* __restrict__ whhb_o) {
    const int NIH = H3 * E_;       // 98304
    const int NHH = H3 * H_;       // 196608
    const int total = 2 * NIH + 2 * NHH;
    for (int i = blockIdx.x * blockDim.x + threadIdx.x; i < total;
         i += gridDim.x * blockDim.x) {
        if (i < NIH)                  wihf_o[i]               = f2bf(wihf[i]);
        else if (i < 2 * NIH)         wihb_o[i - NIH]         = f2bf(wihb[i - NIH]);
        else if (i < 2 * NIH + NHH)   whhf_o[i - 2 * NIH]     = f2bf(whhf[i - 2 * NIH]);
        else                          whhb_o[i - 2*NIH - NHH] = f2bf(whhb[i - 2*NIH - NHH]);
    }
}

// ---------------- kernel 2: fused embedding gather + input projection ----
// xp[dir][t][n][b] = sum_k emb[x[b,t]][k] * w_ih[n][k] + b_ih[n], stored bf16
// in TRANSPOSED layout [T][3H][B] so C-tiles store as one b128 per lane and
// the GRU gate math loads one b128 per gate. One wave per 16x16 tile, K=128.
__global__ __launch_bounds__(128)
void embed_proj_kernel(const int* __restrict__ x,
                       const float* __restrict__ emb,
                       const unsigned short* __restrict__ wih_f,
                       const unsigned short* __restrict__ wih_b,
                       const float* __restrict__ bih_f,
                       const float* __restrict__ bih_b,
                       unsigned short* __restrict__ xp_f,
                       unsigned short* __restrict__ xp_b) {
    const int lane = threadIdx.x & 31;
    const int wave = threadIdx.x >> 5;
    const long tile = (long)blockIdx.x * 4 + wave;     // 8192 * 96 tiles
    const int  mt  = (int)(tile / 96);
    const int  nc  = (int)(tile % 96);
    const int  dir = (nc >= 48) ? 1 : 0;
    const int  nt  = dir ? nc - 48 : nc;

    const unsigned short* wih = dir ? wih_b : wih_f;
    const float*          bih = dir ? bih_b : bih_f;
    unsigned short*       xp  = dir ? xp_b  : xp_f;

    const int l15 = lane & 15;
    const int hif = lane >> 4;             // 0 for lanes 0-15, 1 for 16-31

    // A-row for this lane: flattened m = t*B + b (16-row tile stays in one t)
    const int m = mt * 16 + l15;
    const int t = m >> 8;                  // / B_
    const int b = m & (B_ - 1);
    const int vocab = x[b * T_ + t];
    const float* erow = emb + (long)vocab * E_;

    const int nbase = nt * 16;
    const float bias = bih[nbase + l15];   // C layout: col = l15
    v8f acc = { bias, bias, bias, bias, bias, bias, bias, bias };

    const int aoff = hif * 16;             // A lane K base inside 32-tile
    const int boff = hif * 16;             // B lane K base inside 32-tile

#pragma unroll
    for (int kt = 0; kt < 4; ++kt) {
        const int kb = kt * 32;
        // ---- A tile: gather + native f32->bf16 convert
        V16 a;
        const float4* p0 = (const float4*)(erow + kb + aoff);
        const float4* p1 = (const float4*)(erow + kb + aoff + 8);
        float4 r0 = p0[0], r1 = p0[1], r2 = p1[0], r3 = p1[1];
        a.h[0]  = (__bf16)r0.x; a.h[1]  = (__bf16)r0.y; a.h[2]  = (__bf16)r0.z; a.h[3]  = (__bf16)r0.w;
        a.h[4]  = (__bf16)r1.x; a.h[5]  = (__bf16)r1.y; a.h[6]  = (__bf16)r1.z; a.h[7]  = (__bf16)r1.w;
        a.h[8]  = (__bf16)r2.x; a.h[9]  = (__bf16)r2.y; a.h[10] = (__bf16)r2.z; a.h[11] = (__bf16)r2.w;
        a.h[12] = (__bf16)r3.x; a.h[13] = (__bf16)r3.y; a.h[14] = (__bf16)r3.z; a.h[15] = (__bf16)r3.w;
        // ---- B tile: lane = output column (nbase+l15), contiguous 16 K values
        V16 bm;
        load_btile(bm, wih + (size_t)(nbase + l15) * E_ + kb + boff);
        acc = __builtin_amdgcn_wmma_f32_16x16x32_bf16(
                  false, a.v, false, bm.v, (short)0, acc, false, false);
    }

    // C layout: lane holds rows m = crow0..crow0+7 (consecutive b), col nbase+l15
    const int crow0 = mt * 16 + hif * 8;
    const int tS = crow0 >> 8;
    const int b0 = crow0 & (B_ - 1);
    U4 pk;
    pk.u[0] = pk_bf16(acc[0], acc[1]);
    pk.u[1] = pk_bf16(acc[2], acc[3]);
    pk.u[2] = pk_bf16(acc[4], acc[5]);
    pk.u[3] = pk_bf16(acc[6], acc[7]);
    *(uint4*)(xp + (size_t)tS * (H3 * B_) + (size_t)(nbase + l15) * B_ + b0) = pk.q;
}

// ---------------- kernel 3: GRU scan (one block per direction) ----------
// h kept entirely in LDS as bf16 [B][H] (128 KB of the 320 KB WGP LDS).
// 16 waves; wave j owns output column chunk j (16 cols of H) and computes
// r/z/n for all 16 row tiles each step: 16*3*8 = 384 WMMA per wave per step.
__global__ __launch_bounds__(512)
void gru_kernel(const unsigned short* __restrict__ xp_f,   // [T][3H][B] bf16
                const unsigned short* __restrict__ xp_b,
                const unsigned short* __restrict__ whh_f,  // [3H][H] bf16
                const unsigned short* __restrict__ whh_b,
                const float* __restrict__ bhh_f,
                const float* __restrict__ bhh_b,
                const int* __restrict__ lengths,
                float* __restrict__ hf_out,
                float* __restrict__ hb_out) {
    __shared__ unsigned short hbuf[B_ * H_];          // 131072 bytes
    __shared__ int len_s[B_];                         // 1 KB

    const int dir = blockIdx.x;                       // 0 = fwd, 1 = bwd
    const unsigned short* xp  = dir ? xp_b : xp_f;
    const unsigned short* whh = dir ? whh_b : whh_f;
    const float*          bhh = dir ? bhh_b : bhh_f;
    float*               hout = dir ? hb_out : hf_out;

    for (int i = threadIdx.x; i < B_ * H_; i += 512) hbuf[i] = 0;
    if (threadIdx.x < B_) len_s[threadIdx.x] = lengths[threadIdx.x];
    __syncthreads();

    const int lane = threadIdx.x & 31;
    const int jc   = threadIdx.x >> 5;                // wave id = column chunk
    const int l15  = lane & 15;
    const int hif  = lane >> 4;
    const int nbase = jc * 16;
    const int ccol  = nbase + l15;                    // C/D column of this lane

    const float br = bhh[0 * H_ + ccol];
    const float bz = bhh[1 * H_ + ccol];
    const float bn = bhh[2 * H_ + ccol];

    unsigned hnew_pk[16][4];                          // packed bf16 results

#pragma unroll 1
    for (int step = 0; step < T_; ++step) {
        const int t = dir ? (T_ - 1 - step) : step;
        const unsigned short* xpt = xp + (long)t * (H3 * B_);

        if (step + 1 < T_) {                          // prefetch next xp slice
            const int t2 = dir ? (T_ - 2 - step) : step + 1;
            const unsigned short* pf = xp + (long)t2 * (H3 * B_) + threadIdx.x * 384;
            __builtin_prefetch(pf, 0, 0);
            __builtin_prefetch(pf + 192, 0, 0);
        }

#pragma unroll 1
        for (int rt = 0; rt < 16; ++rt) {
            // ---- A tiles for this row tile straight from LDS (bf16)
            V16 a[8];
            const int arow = rt * 16 + l15;
            const unsigned short* hr = hbuf + arow * H_;
            const int aoff = hif * 16;
#pragma unroll
            for (int kt = 0; kt < 8; ++kt) {
                const uint4* p = (const uint4*)(hr + kt * 32 + aoff);
                const uint4* q = (const uint4*)(hr + kt * 32 + aoff + 8);
                uint4 lo = p[0], hv = q[0];
                a[kt].u32[0] = lo.x; a[kt].u32[1] = lo.y; a[kt].u32[2] = lo.z; a[kt].u32[3] = lo.w;
                a[kt].u32[4] = hv.x; a[kt].u32[5] = hv.y; a[kt].u32[6] = hv.z; a[kt].u32[7] = hv.w;
            }
            v8f accr = { br, br, br, br, br, br, br, br };
            v8f accz = { bz, bz, bz, bz, bz, bz, bz, bz };
            v8f accn = { bn, bn, bn, bn, bn, bn, bn, bn };
#pragma unroll
            for (int kt = 0; kt < 8; ++kt) {
                const int kcol = kt * 32 + hif * 16;  // B lane K base
                V16 b0, b1, b2;
                load_btile(b0, whh + (size_t)(0 * H_ + ccol) * H_ + kcol);
                load_btile(b1, whh + (size_t)(1 * H_ + ccol) * H_ + kcol);
                load_btile(b2, whh + (size_t)(2 * H_ + ccol) * H_ + kcol);
                accr = __builtin_amdgcn_wmma_f32_16x16x32_bf16(
                           false, a[kt].v, false, b0.v, (short)0, accr, false, false);
                accz = __builtin_amdgcn_wmma_f32_16x16x32_bf16(
                           false, a[kt].v, false, b1.v, (short)0, accz, false, false);
                accn = __builtin_amdgcn_wmma_f32_16x16x32_bf16(
                           false, a[kt].v, false, b2.v, (short)0, accn, false, false);
            }
            // ---- gate math (C layout: row = rt*16 + hif*8 + v, col = ccol)
            // xp is [T][3H][B]: one b128 per gate covers rows row0..row0+7.
            const int row0 = rt * 16 + hif * 8;
            U4 xr4, xz4, xn4;
            xr4.q = *(const uint4*)(xpt + (size_t)(0 * H_ + ccol) * B_ + row0);
            xz4.q = *(const uint4*)(xpt + (size_t)(1 * H_ + ccol) * B_ + row0);
            xn4.q = *(const uint4*)(xpt + (size_t)(2 * H_ + ccol) * B_ + row0);
#pragma unroll
            for (int v = 0; v < 8; ++v) {
                const int row = row0 + v;            // row == batch index b
                const float xr = bf2f(xr4.s[v]);
                const float xz = bf2f(xz4.s[v]);
                const float xn = bf2f(xn4.s[v]);
                const float r  = sigmoidf_(xr + accr[v]);
                const float z  = sigmoidf_(xz + accz[v]);
                const float n  = tanhf_(xn + r * accn[v]);
                const unsigned short hold_us = hbuf[row * H_ + ccol];
                const float hold = bf2f(hold_us);
                const float hv = (1.0f - z) * n + z * hold;
                const unsigned short us = (t < len_s[row]) ? f2bf(hv) : hold_us;
                if (v & 1) hnew_pk[rt][v >> 1] |= ((unsigned)us) << 16;
                else       hnew_pk[rt][v >> 1]  = (unsigned)us;
            }
        }
        __syncthreads();                              // all reads of h done
#pragma unroll 1
        for (int rt = 0; rt < 16; ++rt) {
            const int row0 = rt * 16 + hif * 8;
#pragma unroll
            for (int v = 0; v < 8; ++v)
                hbuf[(row0 + v) * H_ + ccol] =
                    (unsigned short)(hnew_pk[rt][v >> 1] >> (16 * (v & 1)));
        }
        __syncthreads();                              // writes visible
    }

    for (int i = threadIdx.x; i < B_ * H_; i += 512)
        hout[i] = bf2f(hbuf[i]);
}

// ---------------- kernel 4: FC head + row normalize ----------------------
__global__ __launch_bounds__(128)
void head_kernel(const float* __restrict__ hf, const float* __restrict__ hb,
                 const float* __restrict__ fc1_w, const float* __restrict__ fc1_b,
                 const float* __restrict__ fc2_w, const float* __restrict__ fc2_b,
                 float* __restrict__ out) {
    __shared__ float hrow[2 * H_];
    __shared__ float hid[128];
    __shared__ float osh[64];
    const int b = blockIdx.x;
    const int j = threadIdx.x;

    for (int i = j; i < H_; i += 128) {
        hrow[i]      = hf[b * H_ + i];
        hrow[H_ + i] = hb[b * H_ + i];
    }
    __syncthreads();

    float s = fc1_b[j];
    for (int k = 0; k < 2 * H_; ++k) s += fc1_w[j * (2 * H_) + k] * hrow[k];
    hid[j] = fmaxf(s, 0.0f);
    __syncthreads();

    float o = 0.0f;
    if (j < 64) {
        o = fc2_b[j];
        for (int k = 0; k < 128; ++k) o += fc2_w[j * 128 + k] * hid[k];
        osh[j] = o * o;
    }
    __syncthreads();
    if (j < 32) osh[j] += osh[j + 32]; __syncthreads();
    if (j < 16) osh[j] += osh[j + 16]; __syncthreads();
    if (j <  8) osh[j] += osh[j +  8]; __syncthreads();
    if (j <  4) osh[j] += osh[j +  4]; __syncthreads();
    if (j <  2) osh[j] += osh[j +  2]; __syncthreads();
    if (j == 0) osh[0] += osh[1];      __syncthreads();
    if (j < 64) {
        const float norm = sqrtf(osh[0]);
        out[b * 64 + j] = o / fmaxf(norm, 1e-12f);
    }
}

// ---------------- host side ----------------------------------------------
static const size_t SZ_XP    = (size_t)T_ * H3 * B_ * 2;       // bf16, [T][3H][B]
static const size_t OFF_XPF  = 0;
static const size_t OFF_XPB  = SZ_XP;
static const size_t OFF_WIHF = 2 * SZ_XP;
static const size_t OFF_WIHB = OFF_WIHF + (size_t)H3 * E_ * 2;
static const size_t OFF_WHHF = OFF_WIHB + (size_t)H3 * E_ * 2;
static const size_t OFF_WHHB = OFF_WHHF + (size_t)H3 * H_ * 2;
static const size_t OFF_HF   = OFF_WHHB + (size_t)H3 * H_ * 2;
static const size_t OFF_HB   = OFF_HF + (size_t)B_ * H_ * 4;

extern "C" void kernel_launch(void* const* d_in, const int* in_sizes, int n_in,
                              void* d_out, int out_size, void* d_ws, size_t ws_size,
                              hipStream_t stream) {
    const int*   x        = (const int*)d_in[0];
    const int*   lengths  = (const int*)d_in[1];
    const float* embedding= (const float*)d_in[2];
    const float* w_ih_f   = (const float*)d_in[3];
    const float* w_hh_f   = (const float*)d_in[4];
    const float* b_ih_f   = (const float*)d_in[5];
    const float* b_hh_f   = (const float*)d_in[6];
    const float* w_ih_b   = (const float*)d_in[7];
    const float* w_hh_b   = (const float*)d_in[8];
    const float* b_ih_b   = (const float*)d_in[9];
    const float* b_hh_b   = (const float*)d_in[10];
    const float* fc1_w    = (const float*)d_in[11];
    const float* fc1_b    = (const float*)d_in[12];
    const float* fc2_w    = (const float*)d_in[13];
    const float* fc2_b    = (const float*)d_in[14];
    float* out = (float*)d_out;

    char* ws = (char*)d_ws;
    unsigned short* xp_f   = (unsigned short*)(ws + OFF_XPF);
    unsigned short* xp_b   = (unsigned short*)(ws + OFF_XPB);
    unsigned short* wihf_o = (unsigned short*)(ws + OFF_WIHF);
    unsigned short* wihb_o = (unsigned short*)(ws + OFF_WIHB);
    unsigned short* whhf_o = (unsigned short*)(ws + OFF_WHHF);
    unsigned short* whhb_o = (unsigned short*)(ws + OFF_WHHB);
    float*          hf     = (float*)(ws + OFF_HF);
    float*          hb     = (float*)(ws + OFF_HB);

    prep_kernel<<<1152, 512, 0, stream>>>(w_ih_f, w_hh_f, w_ih_b, w_hh_b,
                                          wihf_o, whhf_o, wihb_o, whhb_o);

    // 8192 M-tiles * 96 N-tiles (both dirs) / 4 waves per block
    embed_proj_kernel<<<196608, 128, 0, stream>>>(x, embedding,
                                                  wihf_o, wihb_o,
                                                  b_ih_f, b_ih_b,
                                                  xp_f, xp_b);

    gru_kernel<<<2, 512, 0, stream>>>(xp_f, xp_b, whhf_o, whhb_o,
                                      b_hh_f, b_hh_b, lengths, hf, hb);

    head_kernel<<<256, 128, 0, stream>>>(hf, hb, fc1_w, fc1_b, fc2_w, fc2_b, out);
}
